// Decoder_51591147159925
// MI455X (gfx1250) — matrix operand order repaired and checked by reading, not compiled, not run on hardware
//
#include <hip/hip_runtime.h>

// ---------------------------------------------------------------------------
// CDNA5 (gfx1250) wave32 WMMA LSTM captioning decoder.
// GEMMs: v_wmma_f32_16x16x32_bf16.  B tiles staged to LDS with async
// global->LDS copies (ASYNCcnt), double-buffered, shared by 8 waves/WGP.
// Per workgroup: 128 rows x 128 cols (wave w = M-tile w, 8 N-tiles each).
// A fragments pipelined across K-chunks; B fragments pipelined within chunk.
// ---------------------------------------------------------------------------

typedef __attribute__((ext_vector_type(16))) __bf16        v16bf;
typedef __attribute__((ext_vector_type(8)))  float         v8f;
typedef __attribute__((ext_vector_type(4)))  unsigned int  v4u;

union BV { v4u u[2]; v16bf v; };

#define BB   128      // batch
#define TT   32       // timesteps
#define FF   2048     // feature dim
#define EE   512      // embed dim
#define HH   1024     // hidden
#define GG   4096     // 4*H gates
#define VV   10000    // vocab
#define VVP  10240    // vocab padded to 128-col blocks

#define CBLK    128              // output columns per workgroup
#define NT      (CBLK / 16)      // 8 N-tiles per wave
#define RSTRIDE 80               // LDS bytes per staged B row (64B data + pad)
#define BUFB    (CBLK * RSTRIDE) // 10240 B per LDS buffer

// --------------------------- async / wait helpers --------------------------
__device__ __forceinline__ void async_load_b128(unsigned lds_off, const void* gaddr) {
  asm volatile("global_load_async_to_lds_b128 %0, %1, off"
               :: "v"(lds_off), "v"(gaddr)
               : "memory");
}
__device__ __forceinline__ void wait_async_le2() {
  asm volatile("s_wait_asynccnt 0x2" ::: "memory");
}
__device__ __forceinline__ void wait_async_le0() {
  asm volatile("s_wait_asynccnt 0x0" ::: "memory");
}

// --------------------------- fragment loaders ------------------------------
// A-matrix (16x32 bf16, row-major [M,K]):
//   lanes 0-15 : M=lane,    K pairs {0..7,16..23}
//   lanes 16-31: M=lane-16, K pairs {8..15,24..31}
__device__ __forceinline__ v16bf load_a_frag(const __bf16* __restrict__ A,
                                             int lda, int row0, int k0, int lane) {
  int half = lane >> 4;
  int m    = lane & 15;
  const __bf16* p = A + (size_t)(row0 + m) * lda + k0 + 8 * half;
  BV r;
  r.u[0] = *(const v4u*)(p);
  r.u[1] = *(const v4u*)(p + 16);
  return r.v;
}

// B-matrix (32x16) fragment from the LDS-staged tile.
// Staged row r (= output column within block) at byte r*RSTRIDE (64B data).
// lane n = column, half selects K {0..7,16..23} / {8..15,24..31}.
__device__ __forceinline__ v16bf load_b_lds(const v4u* smem, int buf, int nt, int lane) {
  int half = lane >> 4;
  int n    = lane & 15;
  const v4u* p = smem + (buf * BUFB + (nt * 16 + n) * RSTRIDE + half * 16) / 16;
  BV r;
  r.u[0] = p[0];
  r.u[1] = p[2];
  return r.v;
}

// Stage one 128-col x 32-K bf16 chunk of W (row-major [N,K]) into LDS.
// 512 16B items; 256 threads issue 2 asyncs each (items tid and tid+256).
__device__ __forceinline__ void stage_b(const __bf16* __restrict__ W, int ldw,
                                        int col0, int k0, int tid,
                                        unsigned lds_base, int buf) {
#pragma unroll
  for (int it = 0; it < 2; ++it) {
    int i   = tid + it * 256;
    int row = i >> 2, sg = i & 3;
    const __bf16* g = W + (size_t)(col0 + row) * ldw + k0 + sg * 8;
    async_load_b128(lds_base + (unsigned)(buf * BUFB + row * RSTRIDE + sg * 16), g);
  }
}

// One K-chunk of compute: 8 WMMAs, B fragments software-pipelined from LDS.
__device__ __forceinline__ void chunk_mma(const v4u* smem, int buf, int lane,
                                          v16bf a_cur, v8f* acc) {
  v16bf b_cur = load_b_lds(smem, buf, 0, lane);
#pragma unroll
  for (int nt = 0; nt < NT; ++nt) {
    v16bf b_next = b_cur;
    if (nt + 1 < NT) b_next = load_b_lds(smem, buf, nt + 1, lane);
    acc[nt] = __builtin_amdgcn_wmma_f32_16x16x32_bf16(
        false, a_cur, false, b_cur, (short)0, acc[nt], false, false);
    b_cur = b_next;
  }
}

// ------------------------------- kernels -----------------------------------

__global__ void f32_to_bf16_kernel(const float* __restrict__ in,
                                   __bf16* __restrict__ out, int n) {
  int i = blockIdx.x * blockDim.x + threadIdx.x;
  if (i < n) out[i] = (__bf16)in[i];
}

// xs[t][b][e] = bf16(embed_table[captions[b][t]][e])  (PAD row -> 0)
__global__ void embed_kernel(const int* __restrict__ captions,
                             const float* __restrict__ table,
                             __bf16* __restrict__ xs) {
  int idx = blockIdx.x * blockDim.x + threadIdx.x;
  if (idx >= TT * BB * EE) return;
  int e = idx & (EE - 1);
  int b = (idx >> 9) & (BB - 1);
  int t = idx >> 16;               // B*E == 65536
  int cap = captions[b * TT + t];
  float v = (cap == 0) ? 0.0f : table[(size_t)cap * EE + e];
  xs[idx] = (__bf16)v;
}

// init: out[b, j] = features[b,:] . W[j,:] + bias[j];  j = 2*hh + l
// one workgroup: 128 rows x 128 cols; grid.x = 2048/128 = 16
__global__ __launch_bounds__(256)
void init_state_kernel(const __bf16* __restrict__ Fb,   // [128,2048]
                       const __bf16* __restrict__ Wb,   // [2048,2048]
                       const float*  __restrict__ bias, // [2048]
                       float*  __restrict__ fout,       // [2][128][1024] or null
                       __bf16* __restrict__ bout) {     // [2][128][1024] or null
  __shared__ v4u smem[2 * BUFB / 16];
  constexpr int K = FF;
  int tid  = threadIdx.x;
  int wave = tid >> 5, lane = tid & 31;
  int row0 = wave * 16;
  int col0 = blockIdx.x * CBLK;
  unsigned lds_base = (unsigned)(uintptr_t)smem;
  constexpr int total = K / 32;

  v8f acc[NT] = {};
  stage_b(Wb, K, col0, 0, tid, lds_base, 0);
  v16bf a_cur = load_a_frag(Fb, K, row0, 0, lane);
  for (int c = 0; c < total; ++c) {
    int buf = c & 1;
    if (c + 1 < total) { stage_b(Wb, K, col0, 32 * (c + 1), tid, lds_base, buf ^ 1); wait_async_le2(); }
    else               { wait_async_le0(); }
    __syncthreads();
    v16bf a_next = a_cur;
    if (c + 1 < total) a_next = load_a_frag(Fb, K, row0, 32 * (c + 1), lane);
    chunk_mma(smem, buf, lane, a_cur, acc);
    __syncthreads();
    a_cur = a_next;
  }
  int half = lane >> 4, n = lane & 15;
#pragma unroll
  for (int nt = 0; nt < NT; ++nt) {
    int j = col0 + nt * 16 + n;
    int l = j & 1, hh = j >> 1;
#pragma unroll
    for (int r = 0; r < 8; ++r) {
      int b = row0 + r + 8 * half;
      float v = acc[nt][r] + bias[j];
      size_t o = ((size_t)l * BB + b) * HH + hh;
      if (fout) fout[o] = v;
      if (bout) bout[o] = (__bf16)v;
    }
  }
}

// gates[b, 0:4096] = X[b,:].Wih^T + H[b,:].Whh^T + (bih + bhh)
// one workgroup: 128 rows x 128 cols; grid.x = 4096/128 = 32
__global__ __launch_bounds__(256)
void lstm_gates_kernel(const __bf16* __restrict__ X, int Kx,
                       const __bf16* __restrict__ Wih,   // [4096,Kx]
                       const __bf16* __restrict__ Hs,    // [128,1024]
                       const __bf16* __restrict__ Whh,   // [4096,1024]
                       const float* __restrict__ bih,
                       const float* __restrict__ bhh,
                       float* __restrict__ gates) {      // [128,4096]
  __shared__ v4u smem[2 * BUFB / 16];
  int tid  = threadIdx.x;
  int wave = tid >> 5, lane = tid & 31;
  int row0 = wave * 16;
  int col0 = blockIdx.x * CBLK;
  unsigned lds_base = (unsigned)(uintptr_t)smem;
  const int C0 = Kx / 32;
  const int total = C0 + HH / 32;

  auto stage = [&](int c, int buf) {
    if (c < C0) stage_b(Wih, Kx, col0, 32 * c,        tid, lds_base, buf);
    else        stage_b(Whh, HH, col0, 32 * (c - C0), tid, lds_base, buf);
  };
  auto loada = [&](int c) -> v16bf {
    if (c < C0) return load_a_frag(X,  Kx, row0, 32 * c,        lane);
    else        return load_a_frag(Hs, HH, row0, 32 * (c - C0), lane);
  };

  v8f acc[NT] = {};
  stage(0, 0);
  v16bf a_cur = loada(0);
  for (int c = 0; c < total; ++c) {
    int buf = c & 1;
    if (c + 1 < total) { stage(c + 1, buf ^ 1); wait_async_le2(); }
    else               { wait_async_le0(); }
    __syncthreads();
    v16bf a_next = a_cur;
    if (c + 1 < total) a_next = loada(c + 1);
    chunk_mma(smem, buf, lane, a_cur, acc);
    __syncthreads();
    a_cur = a_next;
  }
  int half = lane >> 4, n = lane & 15;
#pragma unroll
  for (int nt = 0; nt < NT; ++nt) {
    int j = col0 + nt * 16 + n;
    float bias = bih[j] + bhh[j];
#pragma unroll
    for (int r = 0; r < 8; ++r) {
      int b = row0 + r + 8 * half;
      gates[(size_t)b * GG + j] = acc[nt][r] + bias;
    }
  }
}

// elementwise LSTM cell; c updated in place (fp32), h emitted as bf16
__global__ void lstm_cell_kernel(const float* __restrict__ gates, // [128,4096]
                                 float* __restrict__ c,           // [128,1024]
                                 __bf16* __restrict__ h_out,      // [128,1024]
                                 __bf16* __restrict__ ys_out) {   // [128,1024] or null
  int idx = blockIdx.x * blockDim.x + threadIdx.x;   // exactly B*H threads
  int b = idx >> 10, hh = idx & (HH - 1);
  const float* g = gates + (size_t)b * GG;
  float gi = g[hh];
  float gf = g[HH + hh];
  float gg = g[2 * HH + hh];
  float go = g[3 * HH + hh];
  float i_ = 1.0f / (1.0f + expf(-gi));
  float f_ = 1.0f / (1.0f + expf(-gf));
  float g_ = tanhf(gg);
  float o_ = 1.0f / (1.0f + expf(-go));
  float cn = f_ * c[idx] + i_ * g_;
  float hn = o_ * tanhf(cn);
  c[idx] = cn;
  h_out[idx] = (__bf16)hn;
  if (ys_out) ys_out[idx] = (__bf16)hn;
}

// out[TB, V] = Ys[TB, H] . fc_w^T + fc_b   (TB = 4096; V padded to 10240)
// grid: x = 10240/128 = 80 col blocks, y = 4096/128 = 32 row groups
__global__ __launch_bounds__(256)
void fc_kernel(const __bf16* __restrict__ Ys,   // [4096,1024]
               const __bf16* __restrict__ Wf,   // [10240,1024] (rows >=10000 garbage)
               const float* __restrict__ bf,
               float* __restrict__ out) {       // [4096,10000]
  __shared__ v4u smem[2 * BUFB / 16];
  int tid  = threadIdx.x;
  int wave = tid >> 5, lane = tid & 31;
  int row0 = blockIdx.y * 128 + wave * 16;
  int col0 = blockIdx.x * CBLK;
  unsigned lds_base = (unsigned)(uintptr_t)smem;
  constexpr int total = HH / 32;

  v8f acc[NT] = {};
  stage_b(Wf, HH, col0, 0, tid, lds_base, 0);
  v16bf a_cur = load_a_frag(Ys, HH, row0, 0, lane);
  for (int c = 0; c < total; ++c) {
    int buf = c & 1;
    if (c + 1 < total) { stage_b(Wf, HH, col0, 32 * (c + 1), tid, lds_base, buf ^ 1); wait_async_le2(); }
    else               { wait_async_le0(); }
    __syncthreads();
    v16bf a_next = a_cur;
    if (c + 1 < total) a_next = load_a_frag(Ys, HH, row0, 32 * (c + 1), lane);
    chunk_mma(smem, buf, lane, a_cur, acc);
    __syncthreads();
    a_cur = a_next;
  }
  int half = lane >> 4, n = lane & 15;
#pragma unroll
  for (int nt = 0; nt < NT; ++nt) {
    int j = col0 + nt * 16 + n;
    if (j < VV) {
      float bias = bf[j];
#pragma unroll
      for (int r = 0; r < 8; ++r) {
        int b = row0 + r + 8 * half;
        out[(size_t)b * VV + j] = acc[nt][r] + bias;
      }
    }
  }
}

// ------------------------------- host --------------------------------------

extern "C" void kernel_launch(void* const* d_in, const int* in_sizes, int n_in,
                              void* d_out, int out_size, void* d_ws, size_t ws_size,
                              hipStream_t stream) {
  const float* features   = (const float*)d_in[0];
  const int*   captions   = (const int*)  d_in[1];
  const float* embed_tab  = (const float*)d_in[2];
  const float* init_h_w   = (const float*)d_in[3];
  const float* init_h_b   = (const float*)d_in[4];
  const float* init_c_w   = (const float*)d_in[5];
  const float* init_c_b   = (const float*)d_in[6];
  const float* w_ih0      = (const float*)d_in[7];
  const float* w_hh0      = (const float*)d_in[8];
  const float* b_ih0      = (const float*)d_in[9];
  const float* b_hh0      = (const float*)d_in[10];
  const float* w_ih1      = (const float*)d_in[11];
  const float* w_hh1      = (const float*)d_in[12];
  const float* b_ih1      = (const float*)d_in[13];
  const float* b_hh1      = (const float*)d_in[14];
  const float* fc_w       = (const float*)d_in[15];
  const float* fc_b       = (const float*)d_in[16];
  float* out = (float*)d_out;

  // ----- workspace carve-up (256B aligned) -----
  char* ws = (char*)d_ws;
  size_t off = 0;
  auto carve = [&](size_t bytes) -> void* {
    void* p = ws + off;
    off = (off + bytes + 255) & ~(size_t)255;
    return p;
  };
  __bf16* wih0_b = (__bf16*)carve((size_t)GG * EE * 2);
  __bf16* whh0_b = (__bf16*)carve((size_t)GG * HH * 2);
  __bf16* wih1_b = (__bf16*)carve((size_t)GG * HH * 2);
  __bf16* whh1_b = (__bf16*)carve((size_t)GG * HH * 2);
  __bf16* fcw_b  = (__bf16*)carve((size_t)VVP * HH * 2);   // padded rows
  __bf16* ihw_b  = (__bf16*)carve((size_t)2 * HH * FF * 2);
  __bf16* icw_b  = (__bf16*)carve((size_t)2 * HH * FF * 2);
  __bf16* feat_b = (__bf16*)carve((size_t)BB * FF * 2);
  __bf16* xs_b   = (__bf16*)carve((size_t)TT * BB * EE * 2);
  __bf16* h_b    = (__bf16*)carve((size_t)2 * BB * HH * 2);   // per-layer h (bf16)
  float*  c_f    = (float*) carve((size_t)2 * BB * HH * 4);   // per-layer c (fp32)
  float*  gates  = (float*) carve((size_t)BB * GG * 4);
  __bf16* ys_b   = (__bf16*)carve((size_t)TT * BB * HH * 2);
  (void)ws_size; (void)in_sizes; (void)n_in; (void)out_size;

  auto cvt = [&](const float* src, __bf16* dst, int n) {
    f32_to_bf16_kernel<<<(n + 255) / 256, 256, 0, stream>>>(src, dst, n);
  };

  // ----- weight conversion to bf16 -----
  cvt(w_ih0,    wih0_b, GG * EE);
  cvt(w_hh0,    whh0_b, GG * HH);
  cvt(w_ih1,    wih1_b, GG * HH);
  cvt(w_hh1,    whh1_b, GG * HH);
  cvt(fc_w,     fcw_b,  VV * HH);
  cvt(init_h_w, ihw_b,  2 * HH * FF);
  cvt(init_c_w, icw_b,  2 * HH * FF);
  cvt(features, feat_b, BB * FF);

  // ----- embedding gather -----
  embed_kernel<<<(TT * BB * EE) / 256, 256, 0, stream>>>(captions, embed_tab, xs_b);

  // ----- initial states -----
  init_state_kernel<<<16, 256, 0, stream>>>(feat_b, ihw_b, init_h_b, nullptr, h_b);
  init_state_kernel<<<16, 256, 0, stream>>>(feat_b, icw_b, init_c_b, c_f, nullptr);

  __bf16* h0 = h_b;
  __bf16* h1 = h_b + (size_t)BB * HH;
  float*  c0 = c_f;
  float*  c1 = c_f + (size_t)BB * HH;

  // ----- sequential LSTM: 32 steps x 2 layers -----
  for (int t = 0; t < TT; ++t) {
    const __bf16* xt = xs_b + (size_t)t * BB * EE;
    lstm_gates_kernel<<<32, 256, 0, stream>>>(xt, EE, wih0_b, h0, whh0_b,
                                              b_ih0, b_hh0, gates);
    lstm_cell_kernel<<<(BB * HH) / 256, 256, 0, stream>>>(gates, c0, h0, nullptr);
    lstm_gates_kernel<<<32, 256, 0, stream>>>(h0, HH, wih1_b, h1, whh1_b,
                                              b_ih1, b_hh1, gates);
    lstm_cell_kernel<<<(BB * HH) / 256, 256, 0, stream>>>(gates, c1, h1,
                                              ys_b + (size_t)t * BB * HH);
  }

  // ----- final FC -----
  fc_kernel<<<dim3(80, 32), 256, 0, stream>>>(ys_b, fcw_b, fc_b, out);
}
